// GCNLayer_41137196761487
// MI455X (gfx1250) — matrix operand order repaired and checked by reading, not compiled, not run on hardware
//
#include <hip/hip_runtime.h>
#include <hip/hip_bf16.h>
#include <cstdint>
#include <cstddef>

#define BLK 256

typedef __attribute__((ext_vector_type(16))) __bf16        v16bf;
typedef __attribute__((ext_vector_type(8)))  float         v8f;
typedef __attribute__((ext_vector_type(4)))  unsigned int  v4u;

static inline long cdivl(long a, long b){ return (a + b - 1) / b; }

__device__ __forceinline__ unsigned short f2bf(float f){
  unsigned u = __float_as_uint(f);
  unsigned r = ((u >> 16) & 1u) + 0x7FFFu;          // round-to-nearest-even
  return (unsigned short)((u + r) >> 16);
}
__device__ __forceinline__ unsigned hash_u32(unsigned v){
  v ^= v >> 17; v *= 0xed5ad4bbu; v ^= v >> 11;
  v *= 0xac4c1b51u; v ^= v >> 15; v *= 0x31848babu; v ^= v >> 14;
  return v;
}
// monotonic float -> uint key (larger float => larger key)
__device__ __forceinline__ unsigned fkey(float f){
  unsigned b = __float_as_uint(f);
  return (b & 0x80000000u) ? ~b : (b | 0x80000000u);
}
__device__ __forceinline__ float fkey_dec(unsigned k){
  unsigned b = (k & 0x80000000u) ? (k ^ 0x80000000u) : ~k;
  return __uint_as_float(b);
}

// ---------------- fills ----------------
__global__ void k_fill_f32(float* p, float v, long n){
  long i = blockIdx.x * (long)blockDim.x + threadIdx.x; if (i < n) p[i] = v;
}
__global__ void k_fill_u32(unsigned* p, unsigned v, long n){
  long i = blockIdx.x * (long)blockDim.x + threadIdx.x; if (i < n) p[i] = v;
}
__global__ void k_fill_i32(int* p, int v, long n){
  long i = blockIdx.x * (long)blockDim.x + threadIdx.x; if (i < n) p[i] = v;
}

// ---------------- bf16 conversion (zero-padded) ----------------
// X [rows,K] f32 row-major -> Xb [rows,Kp] bf16 (K..Kp zero)
__global__ void k_cvt_x(const float* __restrict__ X, unsigned short* __restrict__ Xb,
                        long rows, int K, int Kp){
  long i = blockIdx.x * (long)blockDim.x + threadIdx.x;
  if (i >= rows * (long)Kp) return;
  int  k = (int)(i % Kp);
  long r = i / Kp;
  Xb[i] = (k < K) ? f2bf(X[r * (long)K + k]) : (unsigned short)0;
}
// W [K,Nc] f32 -> Wt [Np,Kp] bf16 transposed+padded (row = output column)
__global__ void k_cvt_w(const float* __restrict__ W, unsigned short* __restrict__ Wt,
                        int K, int Nc, int Kp, int Np){
  int i = blockIdx.x * blockDim.x + threadIdx.x;
  if (i >= Np * Kp) return;
  int k = i % Kp, n = i / Kp;
  Wt[i] = (k < K && n < Nc) ? f2bf(W[(size_t)k * Nc + n]) : (unsigned short)0;
}

// ---------------- WMMA GEMM: C[M,Nout] = A[M,Kp] * Bt[Np,Kp]^T ----------------
// One wave computes a 64(M) x 16(N) block = 4 WMMA tiles sharing one B fragment.
// Fragment layouts per CDNA5 ISA 7.12.2:
//  A (16-bit 16x32): lane l (row=l&15): elems0..7 = K kb..kb+7, elems8..15 = K kb+16..kb+23, kb=(l>>4)*8
//  B (16-bit 32x16): lane l (col=l&15): elems0..7 = K kb..kb+7, elems8..15 = K kb+16..kb+23, kb=(l>>4)*8  (Bt row-major gives contiguity)
#define MT 4
__global__ void k_gemm_bf16(const unsigned short* __restrict__ A,
                            const unsigned short* __restrict__ Bt,
                            float* __restrict__ C,
                            int M, int Kp, int Nout, int Npad){
  long gid  = blockIdx.x * (long)blockDim.x + threadIdx.x;
  long wave = gid >> 5;
  int  lane = (int)(gid & 31);
  int  tilesN  = Npad >> 4;
  long tilesM  = (M + 15) >> 4;
  long tilesM4 = (tilesM + MT - 1) / MT;
  if (wave >= tilesM4 * tilesN) return;
  long tmBase = (wave / tilesN) * MT;
  int  tn     = (int)(wave % tilesN);
  int lr = lane & 15, lh = lane >> 4;

  const unsigned short* ap[MT];
  #pragma unroll
  for (int t = 0; t < MT; t++){
    long row = (tmBase + t) * 16 + lr;
    if (row >= M) row = M - 1;                      // clamp (stores guarded)
    ap[t] = A + row * (long)Kp + lh * 8;
  }
  const unsigned short* bp = Bt + ((size_t)(tn * 16 + lr)) * Kp + lh * 8;

  v8f acc[MT];
  #pragma unroll
  for (int t = 0; t < MT; t++) acc[t] = (v8f){0.f,0.f,0.f,0.f,0.f,0.f,0.f,0.f};

  for (int k = 0; k < Kp; k += 32){
    union { v16bf v; v4u q[2]; } fb;
    fb.q[0] = *(const v4u*)(bp + k);
    fb.q[1] = *(const v4u*)(bp + k + 16);
    #pragma unroll
    for (int t = 0; t < MT; t++){
      __builtin_prefetch((const void*)(ap[t] + k + 32), 0, 1);   // global_prefetch_b8
      union { v16bf v; v4u q[2]; } fa;
      fa.q[0] = *(const v4u*)(ap[t] + k);
      fa.q[1] = *(const v4u*)(ap[t] + k + 16);
      acc[t] = __builtin_amdgcn_wmma_f32_16x16x32_bf16(false, fa.v, false, fb.v,
                                                       (short)0, acc[t], false, false);
    }
  }
  int col = tn * 16 + lr;
  if (col < Nout){
    #pragma unroll
    for (int t = 0; t < MT; t++){
      #pragma unroll
      for (int r = 0; r < 8; r++){
        long row = (tmBase + t) * 16 + r + lh * 8;  // C layout: VGPR r -> M=r / M=r+8
        if (row < M) C[row * (long)Nout + col] = acc[t][r];
      }
    }
  }
}

// ---------------- dual mat-vec (alpha_s / alpha_d), one wave per node ----------------
__global__ void k_matvec2(const float* __restrict__ h, int n, int F,
                          const float* __restrict__ a1, const float* __restrict__ a2,
                          float* __restrict__ o1, float* __restrict__ o2){
  long gid = blockIdx.x * (long)blockDim.x + threadIdx.x;
  long node = gid >> 5; int lane = (int)(gid & 31);
  if (node >= n) return;
  const float* r = h + node * (long)F;
  float s1 = 0.f, s2 = 0.f;
  for (int f = lane; f < F; f += 32){ float v = r[f]; s1 += v * a1[f]; s2 += v * a2[f]; }
  for (int o = 16; o; o >>= 1){ s1 += __shfl_down(s1, o, 32); s2 += __shfl_down(s2, o, 32); }
  if (lane == 0){ o1[node] = s1; o2[node] = s2; }
}

// ---------------- edge softmax: pass A (logits + segment max) ----------------
__global__ void k_edge_logits(const int* __restrict__ src, const int* __restrict__ dst,
                              int E, int Nn,
                              const float* __restrict__ as_, const float* __restrict__ ad_,
                              float* __restrict__ ebuf, unsigned* __restrict__ mkey){
  long i = blockIdx.x * (long)blockDim.x + threadIdx.x;
  if (i >= (long)E + Nn) return;
  int s = (i < E) ? src[i] : (int)(i - E);
  int d = (i < E) ? dst[i] : (int)(i - E);
  float e = as_[s] + ad_[d];
  e = (e > 0.f) ? e : 0.2f * e;            // leaky_relu(0.2)
  ebuf[i] = e;
  atomicMax(&mkey[d], fkey(e));
}
// pass B: exp(e - m[d]) and segment sum
__global__ void k_edge_exp(const int* __restrict__ dst, int E, int Nn,
                           const float* __restrict__ ebuf, const unsigned* __restrict__ mkey,
                           float* __restrict__ denom, float* __restrict__ eebuf){
  long i = blockIdx.x * (long)blockDim.x + threadIdx.x;
  if (i >= (long)E + Nn) return;
  int d = (i < E) ? dst[i] : (int)(i - E);
  float ee = __expf(ebuf[i] - fkey_dec(mkey[d]));
  eebuf[i] = ee;
  atomicAdd(&denom[d], ee);
}
// pass C: out[d] += (ee/denom[d]) * h[s]; one wave per edge
__global__ void k_edge_agg(const int* __restrict__ src, const int* __restrict__ dst,
                           int E, int Nn,
                           const float* __restrict__ eebuf, const float* __restrict__ denom,
                           const float* __restrict__ h, float* __restrict__ out, int F){
  long gid = blockIdx.x * (long)blockDim.x + threadIdx.x;
  long e = gid >> 5; int lane = (int)(gid & 31);
  if (e >= (long)E + Nn) return;
  int s = (e < E) ? src[e] : (int)(e - E);
  int d = (e < E) ? dst[e] : (int)(e - E);
  float w = eebuf[e] / denom[d];
  const float* hs = h + (long)s * F;
  float* od = out + (long)d * F;
  for (int f = lane; f < F; f += 32) atomicAdd(&od[f], w * hs[f]);
}

// ---------------- bias + relu + deterministic dropout (p=0.2) ----------------
__global__ void k_finalize(float* __restrict__ h, const float* __restrict__ bias,
                           long n, int F, unsigned seed){
  long i = blockIdx.x * (long)blockDim.x + threadIdx.x;
  if (i >= n * (long)F) return;
  int f = (int)(i % F);
  float v = h[i] + bias[f];
  v = (v > 0.f) ? v : 0.f;
  unsigned r = hash_u32(seed ^ (unsigned)i);
  h[i] = (r >= 858993459u) ? v * 1.25f : 0.f;   // keep prob 0.8, scale 1/0.8
}

// ---------------- pooling: masked segment sum of h[src] into agg[dst] ----------------
__global__ void k_pool_agg(const int* __restrict__ cs, const int* __restrict__ cd,
                           const float* __restrict__ em,
                           const float* __restrict__ h, float* __restrict__ agg,
                           int E, int F){
  long gid = blockIdx.x * (long)blockDim.x + threadIdx.x;
  long e = gid >> 5; int lane = (int)(gid & 31);
  if (e >= E) return;
  float m = em[e];
  if (m == 0.f) return;
  int s = cs[e], d = cd[e];
  const float* hs = h + (long)s * F;
  float* ad = agg + (long)d * F;
  for (int f = lane; f < F; f += 32) atomicAdd(&ad[f], m * hs[f]);
}
// score = tanh(agg@W_rel + h@W_root + b); one wave per node
__global__ void k_score(const float* __restrict__ agg, const float* __restrict__ h,
                        int n, int F, const float* __restrict__ wrel,
                        const float* __restrict__ wroot, const float* __restrict__ bpool,
                        float* __restrict__ score){
  long gid = blockIdx.x * (long)blockDim.x + threadIdx.x;
  long node = gid >> 5; int lane = (int)(gid & 31);
  if (node >= n) return;
  const float* ar = agg + node * (long)F;
  const float* hr = h   + node * (long)F;
  float s = 0.f;
  for (int f = lane; f < F; f += 32) s += ar[f] * wrel[f] + hr[f] * wroot[f];
  for (int o = 16; o; o >>= 1) s += __shfl_down(s, o, 32);
  if (lane == 0) score[node] = tanhf(s + bpool[0]);
}
// keys: [fkey(score) | ~index]; pad with 0 (minimum)
__global__ void k_build_keys(const float* __restrict__ score,
                             unsigned long long* __restrict__ keys, int n, int P){
  int i = blockIdx.x * blockDim.x + threadIdx.x;
  if (i >= P) return;
  keys[i] = (i < n)
      ? (((unsigned long long)fkey(score[i]) << 32) | (unsigned)(~i))
      : 0ull;
}
// global bitonic compare-swap stage (overall descending)
__global__ void k_bitonic(unsigned long long* __restrict__ key, int P, int size, int stride){
  int t = blockIdx.x * blockDim.x + threadIdx.x;
  if (t >= (P >> 1)) return;
  int idx = ((t / stride) * (stride << 1)) + (t % stride);
  int partner = idx + stride;
  unsigned long long a = key[idx], b = key[partner];
  bool desc = ((idx & size) == 0);
  if (desc ? (a < b) : (a > b)) { key[idx] = b; key[partner] = a; }
}
// gather top-k rows: hout[i] = hin[perm[i]] * score[perm[i]]; inv[perm[i]] = i
__global__ void k_pool_gather(const unsigned long long* __restrict__ keys,
                              const float* __restrict__ score,
                              const float* __restrict__ hin, float* __restrict__ hout,
                              int* __restrict__ inv, int k, int F){
  long gid = blockIdx.x * (long)blockDim.x + threadIdx.x;
  long i = gid >> 5; int lane = (int)(gid & 31);
  if (i >= k) return;
  unsigned idx = ~(unsigned)(keys[i] & 0xFFFFFFFFull);
  float v = score[idx];
  const float* hi = hin + (long)idx * F;
  float* ho = hout + i * (long)F;
  for (int f = lane; f < F; f += 32) ho[f] = hi[f] * v;
  if (lane == 0) inv[idx] = (int)i;
}
__global__ void k_init_edges(const int* __restrict__ src, const int* __restrict__ dst,
                             int* __restrict__ cs, int* __restrict__ cd,
                             float* __restrict__ em, int E){
  int e = blockIdx.x * blockDim.x + threadIdx.x;
  if (e >= E) return;
  cs[e] = src[e]; cd[e] = dst[e]; em[e] = 1.f;
}
__global__ void k_remap(int* __restrict__ cs, int* __restrict__ cd, float* __restrict__ em,
                        const int* __restrict__ inv, int E){
  int e = blockIdx.x * blockDim.x + threadIdx.x;
  if (e >= E) return;
  int ns = inv[cs[e]], nd = inv[cd[e]];
  bool ok = (ns >= 0) && (nd >= 0);
  if (!ok) em[e] = 0.f;
  cs[e] = ok ? ns : 0;
  cd[e] = ok ? nd : 0;
}
__global__ void k_copy_out(const float* __restrict__ h, float* __restrict__ out,
                           int nfin, int F, int OLEN){
  long i = blockIdx.x * (long)blockDim.x + threadIdx.x;
  if (i >= (long)OLEN * F) return;
  int r = (int)(i / F);
  out[i] = (r < nfin) ? h[i] : 0.f;
}

// =====================================================================
extern "C" void kernel_launch(void* const* d_in, const int* in_sizes, int n_in,
                              void* d_out, int out_size, void* d_ws, size_t ws_size,
                              hipStream_t stream) {
  const float* x   = (const float*)d_in[0];
  const int*   src = (const int*)  d_in[1];
  const int*   dst = (const int*)  d_in[2];
  const float* Ws[3] = {(const float*)d_in[3],  (const float*)d_in[7],  (const float*)d_in[11]};
  const float* As[3] = {(const float*)d_in[4],  (const float*)d_in[8],  (const float*)d_in[12]};
  const float* Ad[3] = {(const float*)d_in[5],  (const float*)d_in[9],  (const float*)d_in[13]};
  const float* Bs[3] = {(const float*)d_in[6],  (const float*)d_in[10], (const float*)d_in[14]};
  const float* Wrel  = (const float*)d_in[15];
  const float* Wroot = (const float*)d_in[16];
  const float* bpool = (const float*)d_in[17];

  const int  HD = in_sizes[4];               // 200
  const int  ID = in_sizes[3] / HD;          // 128
  const long N  = (long)in_sizes[0] / ID;    // 50000
  const int  E  = in_sizes[1];               // 800000
  const int  F  = HD;
  const int  KP1 = (ID + 31) / 32 * 32;      // 128
  const int  KP2 = (HD + 31) / 32 * 32;      // 224
  const int  KPm = (KP1 > KP2) ? KP1 : KP2;
  const int  NP  = (HD + 15) / 16 * 16;      // 208
  const int  OLEN = out_size / F;            // 512
  int P0 = 1; while (P0 < N) P0 <<= 1;       // 65536

  // bump allocator over d_ws (256B aligned)
  char* base = (char*)d_ws; size_t off = 0;
  auto alloc = [&](size_t bytes) -> void* {
    void* p = base + off; off = (off + bytes + 255) & ~(size_t)255; return p;
  };
  float*              bufA = (float*)alloc((size_t)N * F * 4);
  float*              bufB = (float*)alloc((size_t)N * F * 4);
  float*              bufC = (float*)alloc((size_t)N * F * 4);   // h_lin / agg
  unsigned short*     xb   = (unsigned short*)alloc((size_t)N * KPm * 2);
  unsigned short*     wb   = (unsigned short*)alloc((size_t)NP * KPm * 2);
  float*              alS  = (float*)alloc(N * 4);
  float*              alD  = (float*)alloc(N * 4);
  unsigned*           mkey = (unsigned*)alloc(N * 4);
  float*              den  = (float*)alloc(N * 4);
  float*              ebuf = (float*)alloc(((size_t)E + N) * 4);
  float*              eeb  = (float*)alloc(((size_t)E + N) * 4);
  int*                cs   = (int*)alloc((size_t)E * 4);
  int*                cd   = (int*)alloc((size_t)E * 4);
  float*              em   = (float*)alloc((size_t)E * 4);
  float*              scr  = (float*)alloc(N * 4);
  unsigned long long* keys = (unsigned long long*)alloc((size_t)P0 * 8);
  int*                inv  = (int*)alloc(N * 4);
  (void)ws_size; (void)n_in;

  const long EN = (long)E + N;

  // -------- 3 GAT layers --------
  for (int L = 0; L < 3; L++){
    const int K  = L ? HD : ID;
    const int KP = L ? KP2 : KP1;
    const float* hin  = (L == 0) ? x : ((L == 1) ? bufA : bufB);
    float*       hacc = (L == 0) ? bufA : ((L == 1) ? bufB : bufA);

    k_cvt_w<<<dim3((unsigned)cdivl((long)NP * KP, BLK)), BLK, 0, stream>>>(Ws[L], wb, K, HD, KP, NP);
    k_cvt_x<<<dim3((unsigned)cdivl(N * KP, BLK)), BLK, 0, stream>>>(hin, xb, N, K, KP);

    long tilesM4 = (((N + 15) / 16) + MT - 1) / MT;
    long waves   = tilesM4 * (NP / 16);
    k_gemm_bf16<<<dim3((unsigned)cdivl(waves * 32, BLK)), BLK, 0, stream>>>(
        xb, wb, bufC, (int)N, KP, F, NP);

    k_matvec2<<<dim3((unsigned)cdivl(N * 32, BLK)), BLK, 0, stream>>>(
        bufC, (int)N, F, As[L], Ad[L], alS, alD);

    k_fill_u32<<<dim3((unsigned)cdivl(N, BLK)), BLK, 0, stream>>>(mkey, 0u, N);
    k_fill_f32<<<dim3((unsigned)cdivl(N, BLK)), BLK, 0, stream>>>(den, 0.f, N);
    k_fill_f32<<<dim3((unsigned)cdivl(N * F, BLK)), BLK, 0, stream>>>(hacc, 0.f, N * (long)F);

    k_edge_logits<<<dim3((unsigned)cdivl(EN, BLK)), BLK, 0, stream>>>(
        src, dst, E, (int)N, alS, alD, ebuf, mkey);
    k_edge_exp<<<dim3((unsigned)cdivl(EN, BLK)), BLK, 0, stream>>>(
        dst, E, (int)N, ebuf, mkey, den, eeb);
    k_edge_agg<<<dim3((unsigned)cdivl(EN * 32, BLK)), BLK, 0, stream>>>(
        src, dst, E, (int)N, eeb, den, bufC, hacc, F);

    k_finalize<<<dim3((unsigned)cdivl(N * F, BLK)), BLK, 0, stream>>>(
        hacc, Bs[L], N, F, 0x1B873593u + (unsigned)L * 0x9E3779B9u);
  }

  // -------- SAGPooling loop --------
  float* h  = bufA;   // after layer 2
  float* hN = bufB;
  float* agg = bufC;
  k_init_edges<<<dim3((unsigned)cdivl(E, BLK)), BLK, 0, stream>>>(src, dst, cs, cd, em, E);

  long n = N;
  while (true){
    k_fill_f32<<<dim3((unsigned)cdivl(n * F, BLK)), BLK, 0, stream>>>(agg, 0.f, n * (long)F);
    k_pool_agg<<<dim3((unsigned)cdivl((long)E * 32, BLK)), BLK, 0, stream>>>(cs, cd, em, h, agg, E, F);
    k_score<<<dim3((unsigned)cdivl(n * 32, BLK)), BLK, 0, stream>>>(
        agg, h, (int)n, F, Wrel, Wroot, bpool, scr);

    int P = 1; while (P < n) P <<= 1;
    k_build_keys<<<dim3((unsigned)cdivl(P, BLK)), BLK, 0, stream>>>(scr, keys, (int)n, P);
    for (int size = 2; size <= P; size <<= 1)
      for (int stride = size >> 1; stride >= 1; stride >>= 1)
        k_bitonic<<<dim3((unsigned)cdivl(P / 2, BLK)), BLK, 0, stream>>>(keys, P, size, stride);

    long kk = (n + 1) / 2;
    k_fill_i32<<<dim3((unsigned)cdivl(n, BLK)), BLK, 0, stream>>>(inv, -1, n);
    k_pool_gather<<<dim3((unsigned)cdivl(kk * 32, BLK)), BLK, 0, stream>>>(
        keys, scr, h, hN, inv, (int)kk, F);
    k_remap<<<dim3((unsigned)cdivl(E, BLK)), BLK, 0, stream>>>(cs, cd, em, inv, E);

    float* t = h; h = hN; hN = t;
    n = kk;
    if (n <= OLEN) break;
  }

  k_copy_out<<<dim3((unsigned)cdivl((long)OLEN * F, BLK)), BLK, 0, stream>>>(
      h, (float*)d_out, (int)n, F, OLEN);
}
